// SpatialEncoder3d_18562848653870
// MI455X (gfx1250) — matrix operand order repaired and checked by reading, not compiled, not run on hardware
//
#include <hip/hip_runtime.h>
#include <math.h>

typedef _Float16 __attribute__((ext_vector_type(16))) v16h;
typedef _Float16 __attribute__((ext_vector_type(8)))  v8h;
typedef float    __attribute__((ext_vector_type(8)))  v8f;

#define B_GRAPHS 8
#define NMAX_    256
#define KDIM     128
#define HDIM     32
#define LDK      136                    // padded row stride in halves (kills bank conflicts)
#define NBLOCKS  1024
#define NTHREADS 256
#define NWAVES   8
#define NTILES   (B_GRAPHS * NMAX_ * (NMAX_ / 16))   // 32768 tiles of 16 pair-rows

__device__ __forceinline__ v16h cat8(v8h lo, v8h hi) {
  return __builtin_shufflevector(lo, hi, 0,1,2,3,4,5,6,7,8,9,10,11,12,13,14,15);
}

__global__ __launch_bounds__(NTHREADS)
void se3d_fused_kernel(const float* __restrict__ coord,
                       const float* __restrict__ gmeans,
                       const float* __restrict__ gstds,
                       const float* __restrict__ mulw,
                       const float* __restrict__ biasw,
                       const float* __restrict__ W1,
                       const float* __restrict__ b1,
                       const float* __restrict__ W2,
                       const float* __restrict__ b2,
                       const int*   __restrict__ ntype,
                       const int*   __restrict__ bnn,
                       float* __restrict__ out,
                       int Ntot)
{
  // W1 transposed: w1t[n*LDK + k] = W1[k][n]   (B-fragments become contiguous per-lane reads)
  __shared__ __attribute__((aligned(16))) _Float16 w1t[KDIM * LDK];
  // W2 transposed: w2t[h*LDK + k] = W2[k][h]
  __shared__ __attribute__((aligned(16))) _Float16 w2t[HDIM * LDK];
  // per-wave staging: 16 rows x 128 K of f16; gk first, then reused for h (gk A-frags live in regs)
  __shared__ __attribute__((aligned(16))) _Float16 astage[NWAVES * 16 * LDK];
  __shared__ float meanv[KDIM], invv[KDIM], coefv[KDIM], b1v[KDIM];
  __shared__ float b2v[HDIM];
  __shared__ int   offs[B_GRAPHS], nbv[B_GRAPHS];

  const int tid = threadIdx.x;

  // ---- one-time per-block setup -------------------------------------------
  for (int idx = tid; idx < KDIM * KDIM; idx += NTHREADS) {
    int k = idx >> 7, n = idx & 127;
    w1t[n * LDK + k] = (_Float16)W1[idx];
  }
  for (int idx = tid; idx < KDIM * HDIM; idx += NTHREADS) {
    int k = idx >> 5, h = idx & 31;
    w2t[h * LDK + k] = (_Float16)W2[idx];
  }
  if (tid < KDIM) {
    float s   = fabsf(gstds[tid]) + 0.01f;
    float inv = 1.0f / s;
    meanv[tid] = gmeans[tid];
    invv[tid]  = inv;
    coefv[tid] = -0.39894228040143267794f * inv;   // -1/(sqrt(2*pi)*std)
    b1v[tid]   = b1[tid];
  }
  if (tid < HDIM) b2v[tid] = b2[tid];
  if (tid == 0) {
    int acc = 0;
    for (int g = 0; g < B_GRAPHS; ++g) { offs[g] = acc; int nb = bnn[g]; nbv[g] = nb; acc += nb; }
  }
  __syncthreads();

  const int lane  = tid & 31;
  const int widx  = tid >> 5;
  _Float16* ast   = &astage[widx * 16 * LDK];
  const int r     = lane & 15;            // A row / C column within 16-tile
  const int hiL   = lane >> 4;            // lane half
  const int kbA   = hiL ? 8 : 0;          // A-frag: lanes 0-15 K{0..7,16..23}, 16-31 K{8..15,24..31}
  const int kbB   = hiL ? 8 : 0;          // B-frag: lanes 0-15 K{0..7,16..23}, 16-31 K{8..15,24..31}
  const int rbase = hiL ? 8 : 0;          // C-frag: VGPR v -> row v (lo lanes) / v+8 (hi lanes)

  const int gwave = blockIdx.x * NWAVES + widx;

  for (int tile = gwave; tile < NTILES; tile += NBLOCKS * NWAVES) {
    const int b   = tile >> 12;
    const int i   = (tile >> 4) & 255;
    const int j0  = (tile & 15) << 4;
    const int off = offs[b];
    const int nb  = nbv[b];

    // ---- per-row scaled distance (row = j0 + r), clamped like the reference
    const int j    = j0 + r;
    const int iidx = min(off + i, Ntot - 1);
    const int jidx = min(off + j, Ntot - 1);
    float xi = coord[3*iidx+0], yi = coord[3*iidx+1], zi = coord[3*iidx+2];
    float xj = coord[3*jidx+0], yj = coord[3*jidx+1], zj = coord[3*jidx+2];
    float dx = xi - xj, dy = yi - yj, dz = zi - zj;
    float sq = dx*dx + dy*dy + dz*dz;
    float dist = sq > 0.0f ? sqrtf(sq) : 0.0f;
    int ti = ntype[iidx], tj = ntype[jidx];
    float mul = mulw[2*ti+0]  + mulw[2*tj+1];
    float bia = biasw[2*ti+0] + biasw[2*tj+1];
    float scaled = fmaf(mul, dist, bia);

    // ---- Gaussian RBF features -> f16 staging [16 x 128] (2 lanes/row, 64 k each)
    const int k0 = hiL * 64;
    #pragma unroll
    for (int blk = 0; blk < 8; ++blk) {
      v8h pk;
      #pragma unroll
      for (int t = 0; t < 8; ++t) {
        int k = k0 + blk * 8 + t;
        float z = (scaled - meanv[k]) * invv[k];
        float g = __expf(-0.5f * z * z) * coefv[k];
        pk[t] = (_Float16)g;
      }
      *(v8h*)&ast[r * LDK + k0 + blk * 8] = pk;
    }
    asm volatile("" ::: "memory");   // wave-private LDS; DS ops are in-order within a wave

    // ---- A fragments (gk) into registers, kept across all N-tiles
    v16h afr[4];
    #pragma unroll
    for (int q = 0; q < 4; ++q) {
      int kc = q * 32;
      v8h lo = *(const v8h*)&ast[r * LDK + kc + kbA];
      v8h hi = *(const v8h*)&ast[r * LDK + kc + kbA + 16];
      afr[q] = cat8(lo, hi);
    }
    asm volatile("" ::: "memory");

    // ---- GEMM1 (16x128 @ 128x128) + bias + exact GELU, h overwrites staging
    #pragma unroll
    for (int nt = 0; nt < 8; ++nt) {
      int n0 = nt * 16;
      v8f acc = {};
      #pragma unroll
      for (int q = 0; q < 4; ++q) {
        int kc = q * 32;
        const _Float16* wp = &w1t[(n0 + r) * LDK + kc + kbB];
        v16h bfr = cat8(*(const v8h*)wp, *(const v8h*)(wp + 16));
        acc = __builtin_amdgcn_wmma_f32_16x16x32_f16(false, afr[q], false, bfr,
                                                     (short)0, acc, false, false);
      }
      float bb = b1v[n0 + r];
      #pragma unroll
      for (int v = 0; v < 8; ++v) {
        float x = acc[v] + bb;
        float g = 0.5f * x * (1.0f + erff(x * 0.70710678118654752440f)); // exact gelu
        ast[(v + rbase) * LDK + n0 + r] = (_Float16)g;
      }
    }
    asm volatile("" ::: "memory");

    // ---- h fragments
    v16h hfr[4];
    #pragma unroll
    for (int q = 0; q < 4; ++q) {
      int kc = q * 32;
      v8h lo = *(const v8h*)&ast[r * LDK + kc + kbA];
      v8h hi = *(const v8h*)&ast[r * LDK + kc + kbA + 16];
      hfr[q] = cat8(lo, hi);
    }

    // ---- GEMM2 (16x128 @ 128x32) + bias + validity mask + store
    const bool vi = (i < nb);
    const size_t rowbase = (size_t)(b * NMAX_ + i) * NMAX_;
    #pragma unroll
    for (int nt = 0; nt < 2; ++nt) {
      int n0 = nt * 16;
      v8f acc = {};
      #pragma unroll
      for (int q = 0; q < 4; ++q) {
        int kc = q * 32;
        const _Float16* wp = &w2t[(n0 + r) * LDK + kc + kbB];
        v16h bfr = cat8(*(const v8h*)wp, *(const v8h*)(wp + 16));
        acc = __builtin_amdgcn_wmma_f32_16x16x32_f16(false, hfr[q], false, bfr,
                                                     (short)0, acc, false, false);
      }
      float bb = b2v[n0 + r];
      #pragma unroll
      for (int v = 0; v < 8; ++v) {
        int jj = j0 + v + rbase;                       // pair row of this C element
        float val = (vi && (jj - j0 + j0) < nb && jj < nb) ? (acc[v] + bb) : -INFINITY;
        out[(rowbase + (size_t)jj) * HDIM + n0 + r] = val;
      }
    }
    asm volatile("" ::: "memory");
  }
}

extern "C" void kernel_launch(void* const* d_in, const int* in_sizes, int n_in,
                              void* d_out, int out_size, void* d_ws, size_t ws_size,
                              hipStream_t stream) {
  const float* coord  = (const float*)d_in[0];
  const float* gmeans = (const float*)d_in[1];
  const float* gstds  = (const float*)d_in[2];
  const float* mulw   = (const float*)d_in[3];
  const float* biasw  = (const float*)d_in[4];
  const float* W1     = (const float*)d_in[5];
  const float* b1     = (const float*)d_in[6];
  const float* W2     = (const float*)d_in[7];
  const float* b2     = (const float*)d_in[8];
  const int*   ntype  = (const int*)d_in[9];
  const int*   bnn    = (const int*)d_in[10];
  float* out = (float*)d_out;
  int Ntot = in_sizes[0] / 3;

  hipLaunchKernelGGL(se3d_fused_kernel, dim3(NBLOCKS), dim3(NTHREADS), 0, stream,
                     coord, gmeans, gstds, mulw, biasw, W1, b1, W2, b2,
                     ntype, bnn, out, Ntot);
}
